// MomentumVectorQuantizer_15839839387914
// MI455X (gfx1250) — matrix operand order repaired and checked by reading, not compiled, not run on hardware
//
#include <hip/hip_runtime.h>
#include <hip/hip_bf16.h>

typedef __attribute__((ext_vector_type(16))) _Float16 v16h;
typedef __attribute__((ext_vector_type(8)))  float    v8f;
typedef __attribute__((ext_vector_type(4)))  _Float16 h4;

#define C_DIM 512   // feature dim (K of the GEMM)

// ---------------------------------------------------------------------------
// Kernel 1: L2-normalize each token row of x, emit f16 row-major [N][512].
// One wave32 per row; lane handles 16 floats as 4 coalesced float4 loads.
// ---------------------------------------------------------------------------
__global__ __launch_bounds__(256) void k_norm_x(const float* __restrict__ x,
                                                _Float16* __restrict__ xn,
                                                int nrows) {
  int w = threadIdx.x >> 5, lane = threadIdx.x & 31;
  int row = blockIdx.x * 8 + w;
  if (row >= nrows) return;
  const float* p = x + (size_t)row * C_DIM;
  float4 v[4];
  float s = 0.f;
#pragma unroll
  for (int i = 0; i < 4; ++i) {
    v[i] = *(const float4*)(p + i * 128 + lane * 4);
    s += v[i].x * v[i].x + v[i].y * v[i].y + v[i].z * v[i].z + v[i].w * v[i].w;
  }
#pragma unroll
  for (int m = 16; m >= 1; m >>= 1) s += __shfl_xor(s, m, 32);
  float scale = 1.0f / fmaxf(sqrtf(s), 1e-12f);
  _Float16* q = xn + (size_t)row * C_DIM;
#pragma unroll
  for (int i = 0; i < 4; ++i) {
    h4 h;
    h[0] = (_Float16)(v[i].x * scale);
    h[1] = (_Float16)(v[i].y * scale);
    h[2] = (_Float16)(v[i].z * scale);
    h[3] = (_Float16)(v[i].w * scale);
    *(h4*)(q + i * 128 + lane * 4) = h;
  }
}

// ---------------------------------------------------------------------------
// Kernel 2: L2-normalize each codebook column of emb[C][M]; write
//  - eh : f16  [M][512] column-major (B-matrix feed for WMMA)
//  - et : f32  [M][512] transpose    (for the final gather)
// ---------------------------------------------------------------------------
__global__ __launch_bounds__(256) void k_norm_emb(const float* __restrict__ emb,
                                                  _Float16* __restrict__ eh,
                                                  float* __restrict__ et,
                                                  int mcols) {
  int col = blockIdx.x * 256 + threadIdx.x;
  if (col >= mcols) return;
  float s = 0.f;
  for (int k = 0; k < C_DIM; ++k) {
    float v = emb[(size_t)k * mcols + col];
    s += v * v;
  }
  float scale = 1.0f / fmaxf(sqrtf(s), 1e-12f);
  for (int k = 0; k < C_DIM; ++k) {
    float v = emb[(size_t)k * mcols + col] * scale;
    et[(size_t)col * C_DIM + k] = v;
    eh[(size_t)col * C_DIM + k] = (_Float16)v;
  }
}

// ---------------------------------------------------------------------------
// Kernel 3: fused similarity GEMM (f16 WMMA, f32 accum) + per-row argmax.
// Block = 256 threads = 8 waves (2 row-tiles x 4 col-pairs); 32 token rows
// per block.  A panel (32x512 f16 = 32KB) is staged into LDS with async
// global->LDS loads (ASYNCcnt), then each wave pins its 16x512 A panel in
// registers.  B is streamed with a register triple-buffer; sched_group
// barriers force [4 loads | 2 WMMA] interleaving so each WMMA's operands
// were loaded two stages earlier (waits become loadcnt<=8, not 0).
// ---------------------------------------------------------------------------
__global__ __launch_bounds__(256) void k_gemm_argmax(
    const _Float16* __restrict__ xn, const _Float16* __restrict__ eh,
    int* __restrict__ best, int mcols) {
  __shared__ _Float16 lds_a[32 * C_DIM];   // 32 KB A panel

  int tid = threadIdx.x;
  int w = tid >> 5, lane = tid & 31;
  int rowtile = w >> 2;      // 0..1
  int ct = w & 3;            // 0..3 -> which 32-column pair
  int nb = blockIdx.x * 32 + rowtile * 16;
  int lrow = lane & 15;
  int lhalf = lane >> 4;     // 0: low K half, 1: high K half

  // ---- Async-stage the block's A panel into LDS (8 x 4KB = 32KB).
  {
    uint32_t lbase = (uint32_t)(uintptr_t)lds_a;        // LDS byte offset
    uint64_t gbase = (uint64_t)(uintptr_t)(xn + (size_t)blockIdx.x * 32 * C_DIM);
#pragma unroll
    for (int j = 0; j < 8; ++j) {
      uint32_t loff = lbase + (uint32_t)tid * 16u + (uint32_t)j * 4096u;
      uint64_t goff = gbase + (uint64_t)tid * 16u + (uint64_t)j * 4096u;
      asm volatile("global_load_async_to_lds_b128 %0, %1, off"
                   :: "v"(loff), "v"(goff) : "memory");
    }
    asm volatile("s_wait_asynccnt 0x0" ::: "memory");
    __syncthreads();
  }

  // ---- A fragments from LDS: row (rowtile*16+lrow), K chunk kf*32.
  // Layout: lanes 0-15 hold K = 0..15, lanes 16-31 hold K = 16..31 (matches B).
  v16h af[16];
  const _Float16* la = lds_a + (size_t)(rowtile * 16 + lrow) * C_DIM + lhalf * 16;
#pragma unroll
  for (int kf = 0; kf < 16; ++kf) {
    const _Float16* p = la + kf * 32;
    ((uint4*)&af[kf])[0] = *(const uint4*)(p);        // halves 0..7
    ((uint4*)&af[kf])[1] = *(const uint4*)(p + 8);    // halves 8..15
  }

  float bestv[8];
  int   besti[8];
#pragma unroll
  for (int r = 0; r < 8; ++r) { bestv[r] = -3.0e38f; besti[r] = 0; }

  const int niter = mcols >> 7;  // 128 columns per block-iteration
  // B lane feed: lane n<16 -> column n, K = kf*32 + 0..15 contiguous;
  // lanes 16-31 -> same column, K = kf*32 + 16..31.
  const _Float16* bb = eh + (size_t)(ct * 32 + lrow) * C_DIM + lhalf * 16;

  for (int it = 0; it < niter; ++it) {
    const _Float16* p0 = bb + (size_t)it * 128 * C_DIM;       // col tile 0
    const _Float16* p1 = p0 + (size_t)16 * C_DIM;             // col tile 1

    v8f acc0 = {0.f, 0.f, 0.f, 0.f, 0.f, 0.f, 0.f, 0.f};
    v8f acc1 = {0.f, 0.f, 0.f, 0.f, 0.f, 0.f, 0.f, 0.f};

    // Register triple-buffers for the two B streams (8 loads in flight).
    v16h b0[3], b1[3];
#pragma unroll
    for (int s = 0; s < 2; ++s) {                 // preload stages 0,1
      const _Float16* q0 = p0 + s * 32;
      const _Float16* q1 = p1 + s * 32;
      ((uint4*)&b0[s])[0] = *(const uint4*)(q0);
      ((uint4*)&b0[s])[1] = *(const uint4*)(q0 + 8);
      ((uint4*)&b1[s])[0] = *(const uint4*)(q1);
      ((uint4*)&b1[s])[1] = *(const uint4*)(q1 + 8);
    }

#pragma unroll
    for (int kf = 0; kf < 16; ++kf) {
      if (kf < 14) {                              // fetch stage kf+2
        int s = (kf + 2) % 3;
        const _Float16* q0 = p0 + (kf + 2) * 32;
        const _Float16* q1 = p1 + (kf + 2) * 32;
        ((uint4*)&b0[s])[0] = *(const uint4*)(q0);
        ((uint4*)&b0[s])[1] = *(const uint4*)(q0 + 8);
        ((uint4*)&b1[s])[0] = *(const uint4*)(q1);
        ((uint4*)&b1[s])[1] = *(const uint4*)(q1 + 8);
      }
      int c = kf % 3;
      // Two independent accumulation chains sharing the A fragment.
      acc0 = __builtin_amdgcn_wmma_f32_16x16x32_f16(
          false, af[kf], false, b0[c], (short)0, acc0, false, false);
      acc1 = __builtin_amdgcn_wmma_f32_16x16x32_f16(
          false, af[kf], false, b1[c], (short)0, acc1, false, false);
    }

    // Enforce software-pipelined schedule:
    //   VMEM(8) ; 14 x [VMEM(4), WMMA(2)] ; WMMA(2) ; WMMA(2)
    // 0x020 = VMEM read group, 0x008 = MFMA/WMMA group.
    __builtin_amdgcn_sched_group_barrier(0x020, 8, 0);
#pragma unroll
    for (int g = 0; g < 14; ++g) {
      __builtin_amdgcn_sched_group_barrier(0x020, 4, 0);
      __builtin_amdgcn_sched_group_barrier(0x008, 2, 0);
    }
    __builtin_amdgcn_sched_group_barrier(0x008, 2, 0);
    __builtin_amdgcn_sched_group_barrier(0x008, 2, 0);

    // C layout: VGPR r, lanes 0-15 -> row nb+r, lanes 16-31 -> row nb+8+r;
    // tile-0 column = it*128 + ct*32 + lrow, tile-1 column = +16.
    int col0 = it * 128 + ct * 32 + lrow;
    int col1 = col0 + 16;
#pragma unroll
    for (int r = 0; r < 8; ++r) {
      if (acc0[r] > bestv[r]) { bestv[r] = acc0[r]; besti[r] = col0; }
      if (acc1[r] > bestv[r] || (acc1[r] == bestv[r] && col1 < besti[r])) {
        bestv[r] = acc1[r]; besti[r] = col1;
      }
    }
  }

  // Argmax reduction across the 16 lanes sharing each row (masks 1..8 never
  // cross the 16-lane halves). Tie-break on lowest index (first occurrence).
#pragma unroll
  for (int m = 1; m <= 8; m <<= 1) {
#pragma unroll
    for (int r = 0; r < 8; ++r) {
      float ov = __shfl_xor(bestv[r], m, 32);
      int   oi = __shfl_xor(besti[r], m, 32);
      if (ov > bestv[r] || (ov == bestv[r] && oi < besti[r])) {
        bestv[r] = ov; besti[r] = oi;
      }
    }
  }
  if (lane == 0) {
#pragma unroll
    for (int r = 0; r < 8; ++r) best[nb + r] = besti[r];
  } else if (lane == 16) {
#pragma unroll
    for (int r = 0; r < 8; ++r) best[nb + 8 + r] = besti[r];
  }
}

// ---------------------------------------------------------------------------
// Kernel 4: out[n][:] = embT_f32[best[n]][:]  (contiguous 2KB row copies)
// ---------------------------------------------------------------------------
__global__ __launch_bounds__(256) void k_gather(const int* __restrict__ best,
                                                const float* __restrict__ et,
                                                float* __restrict__ out,
                                                int nrows) {
  int row = blockIdx.x * 2 + (threadIdx.x >> 7);
  int c4 = threadIdx.x & 127;
  if (row >= nrows) return;
  int idx = best[row];
  float4 v = *(const float4*)(et + (size_t)idx * C_DIM + c4 * 4);
  *(float4*)(out + (size_t)row * C_DIM + c4 * 4) = v;
}

extern "C" void kernel_launch(void* const* d_in, const int* in_sizes, int n_in,
                              void* d_out, int out_size, void* d_ws, size_t ws_size,
                              hipStream_t stream) {
  const float* x   = (const float*)d_in[0];   // [8,4096,512] f32
  const float* emb = (const float*)d_in[1];   // [512, M]     f32
  int N = in_sizes[0] / C_DIM;                // 32768 tokens
  int M = in_sizes[1] / C_DIM;                // 8192 codes
  float* out = (float*)d_out;                 // [N,512] f32

  // workspace layout
  char* ws = (char*)d_ws;
  _Float16* xn = (_Float16*)ws;                                  // N*512*2 B
  size_t off = (size_t)N * C_DIM * 2;
  _Float16* eh = (_Float16*)(ws + off);                          // M*512*2 B
  off += (size_t)M * C_DIM * 2;
  float* et = (float*)(ws + off);                                // M*512*4 B
  off += (size_t)M * C_DIM * 4;
  int* best = (int*)(ws + off);                                  // N*4 B

  k_norm_x<<<N / 8, 256, 0, stream>>>(x, xn, N);
  k_norm_emb<<<(M + 255) / 256, 256, 0, stream>>>(emb, eh, et, M);
  k_gemm_argmax<<<N / 32, 256, 0, stream>>>(xn, eh, best, M);
  k_gather<<<N / 2, 256, 0, stream>>>(best, et, out, N);
}
